// TransformerLayer_37022618092047
// MI455X (gfx1250) — compile-verified, matrix operand order checked by reading, not run on hardware
//
#include <hip/hip_runtime.h>
#include <hip/hip_bf16.h>

#define B_  2
#define N_  256
#define D_  256
#define E_  256
#define TM_ 128   // edge rows per block in the fused edge kernel

typedef __attribute__((ext_vector_type(16))) __bf16 v16bf;
typedef __attribute__((ext_vector_type(8)))  __bf16 v8bf;
typedef __attribute__((ext_vector_type(8)))  float  v8f;

__device__ __forceinline__ v8f zero_v8f() {
    v8f z;
#pragma unroll
    for (int u = 0; u < 8; ++u) z[u] = 0.f;
    return z;
}

__device__ __forceinline__ v8f wmma_bf(v16bf a, v16bf b, v8f c) {
    return __builtin_amdgcn_wmma_f32_16x16x32_bf16(false, a, false, b, (short)0, c, false, false);
}

// Load a 16x32 bf16 A-fragment (or the B-fragment of W^T from row-major W).
// lane L (<16): row = L, K elems {0..7} and {16..23} (+8 for upper lane half).
__device__ __forceinline__ v16bf load_frag(const __bf16* base, int stride) {
    const int lane = threadIdx.x & 31;
    const __bf16* p = base + (size_t)(lane & 15) * stride + ((lane >> 4) << 3);
    v8bf lo = *(const v8bf*)p;
    v8bf hi = *(const v8bf*)(p + 16);
    return __builtin_shufflevector(lo, hi, 0, 1, 2, 3, 4, 5, 6, 7, 8, 9, 10, 11, 12, 13, 14, 15);
}

// ---------------- K0: f32 -> bf16 weight conversion ----------------
__global__ void k_convert(const float* __restrict__ Weo, const float* __restrict__ We1,
                          const float* __restrict__ We2, __bf16* __restrict__ Weo_bf,
                          __bf16* __restrict__ We1_bf, __bf16* __restrict__ We2_bf) {
    size_t idx = (size_t)blockIdx.x * 256 + threadIdx.x;
    if (idx < 65536) {
        Weo_bf[idx] = (__bf16)Weo[idx];
    } else if (idx < 65536 + 262144) {
        We1_bf[idx - 65536] = (__bf16)We1[idx - 65536];
    } else if (idx < 65536 + 524288) {
        We2_bf[idx - 327680] = (__bf16)We2[idx - 327680];
    }
}

// ---------------- K1: x = nodes + conds; qkv projection ----------------
__global__ __launch_bounds__(256) void k_qkv(const float* __restrict__ nodes,
                                             const float* __restrict__ conds,
                                             const float* __restrict__ Wqkv,
                                             const float* __restrict__ bqkv,
                                             float* __restrict__ q, float* __restrict__ k,
                                             float* __restrict__ v) {
    const int bn = blockIdx.x;        // b*N + n
    const int b  = bn >> 8;
    const int t  = threadIdx.x;
    __shared__ float xrow[D_];
    xrow[t] = nodes[(size_t)bn * D_ + t] + conds[b * D_ + t];
    __syncthreads();
#pragma unroll
    for (int p = 0; p < 3; ++p) {
        const int f = p * 256 + t;
        const float* wr = Wqkv + (size_t)f * D_;
        float acc = bqkv[f];
        for (int c = 0; c < D_; ++c) acc += xrow[c] * wr[c];
        float* dst = (p == 0) ? q : ((p == 1) ? k : v);
        dst[(size_t)bn * D_ + t] = acc;
    }
}

// ---------------- K2: edge-biased attention -> wv ----------------
__global__ __launch_bounds__(256) void k_attn(const float* __restrict__ edges,
                                              const float* __restrict__ qb,
                                              const float* __restrict__ kb,
                                              const float* __restrict__ vb,
                                              float* __restrict__ wv) {
    const int bn = blockIdx.x;        // b*N + i
    const int b  = bn >> 8;
    const int t = threadIdx.x, lane = t & 31, wid = t >> 5;
    __shared__ float qrow[D_];
    __shared__ float logits[8][N_];
    qrow[t] = qb[(size_t)bn * D_ + t];
    __syncthreads();
    const float inv = 0.0625f;  // 1/sqrt(256)
    for (int jj = 0; jj < 32; ++jj) {
        const int j = wid * 32 + jj;
        const float* er = edges + ((size_t)bn * N_ + j) * E_;
        const float* kr = kb + ((size_t)(b * N_ + j)) * D_;
        float part[8];
#pragma unroll
        for (int h = 0; h < 8; ++h) {
            const int c = h * 32 + lane;
            part[h] = er[c] + qrow[c] * kr[c] * inv;
        }
#pragma unroll
        for (int h = 0; h < 8; ++h) {
            float x = part[h];
#pragma unroll
            for (int off = 16; off; off >>= 1) x += __shfl_xor(x, off, 32);
            part[h] = x;
        }
        if (lane == 0) {
#pragma unroll
            for (int h = 0; h < 8; ++h) logits[h][j] = part[h];
        }
    }
    __syncthreads();
    // softmax over j; wave 'wid' owns head 'wid'
    {
        float vals[8];
        float mx = -3.0e38f;
#pragma unroll
        for (int u = 0; u < 8; ++u) {
            vals[u] = logits[wid][lane + 32 * u];
            mx = fmaxf(mx, vals[u]);
        }
#pragma unroll
        for (int off = 16; off; off >>= 1) mx = fmaxf(mx, __shfl_xor(mx, off, 32));
        float sum = 0.f;
#pragma unroll
        for (int u = 0; u < 8; ++u) {
            vals[u] = __expf(vals[u] - mx);
            sum += vals[u];
        }
#pragma unroll
        for (int off = 16; off; off >>= 1) sum += __shfl_xor(sum, off, 32);
        const float isum = 1.f / sum;
#pragma unroll
        for (int u = 0; u < 8; ++u) logits[wid][lane + 32 * u] = vals[u] * isum;
    }
    __syncthreads();
    // wv[i, f] = sum_j attn[h][j] * v[j, f]   (f = t, h = t>>5)
    float acc = 0.f;
    for (int j = 0; j < N_; ++j) acc += logits[wid][j] * vb[((size_t)(b * N_ + j)) * D_ + t];
    wv[(size_t)bn * D_ + t] = acc;
}

// ---------------- helpers: block reduction for node LN ----------------
__device__ __forceinline__ float blk_sum(float x, float* red) {
#pragma unroll
    for (int off = 16; off; off >>= 1) x += __shfl_xor(x, off, 32);
    const int t = threadIdx.x;
    if ((t & 31) == 0) red[t >> 5] = x;
    __syncthreads();
    float s = red[0] + red[1] + red[2] + red[3] + red[4] + red[5] + red[6] + red[7];
    __syncthreads();
    return s;
}

// ---------------- K3: node out-proj + residual + LN + MLP + LN ----------------
__global__ __launch_bounds__(256) void k_nodes(const float* __restrict__ wv,
                                               const float* __restrict__ nodes,
                                               const float* __restrict__ Wno,
                                               const float* __restrict__ bno,
                                               const float* __restrict__ g1n,
                                               const float* __restrict__ b1n,
                                               const float* __restrict__ Wn1,
                                               const float* __restrict__ bn1,
                                               const float* __restrict__ Wn2,
                                               const float* __restrict__ bn2,
                                               const float* __restrict__ g2n,
                                               const float* __restrict__ b2n,
                                               float* __restrict__ out_nodes) {
    const int bn = blockIdx.x;
    const int t = threadIdx.x;
    __shared__ float wrow[D_], xrow[D_], hid[4 * D_];
    __shared__ float red[8];
    wrow[t] = wv[(size_t)bn * D_ + t];
    __syncthreads();
    const float* wr = Wno + (size_t)t * D_;
    float acc = bno[t];
    for (int c = 0; c < D_; ++c) acc += wrow[c] * wr[c];
    float val = acc + nodes[(size_t)bn * D_ + t];
    float s = blk_sum(val, red);
    float m = s * (1.f / 256.f);
    float d0 = val - m;
    float vs = blk_sum(d0 * d0, red) * (1.f / 256.f);
    float xn = d0 * rsqrtf(vs + 1e-5f) * g1n[t] + b1n[t];
    xrow[t] = xn;
    __syncthreads();
#pragma unroll
    for (int u = 0; u < 4; ++u) {
        const int hf = u * 256 + t;
        const float* w1 = Wn1 + (size_t)hf * D_;
        float h = bn1[hf];
        for (int c = 0; c < D_; ++c) h += xrow[c] * w1[c];
        hid[hf] = fmaxf(h, 0.f);
    }
    __syncthreads();
    const float* w2 = Wn2 + (size_t)t * 1024;
    float o = bn2[t];
    for (int c = 0; c < 1024; ++c) o += hid[c] * w2[c];
    o += xn;
    s = blk_sum(o, red);
    m = s * (1.f / 256.f);
    d0 = o - m;
    vs = blk_sum(d0 * d0, red) * (1.f / 256.f);
    out_nodes[(size_t)bn * D_ + t] = d0 * rsqrtf(vs + 1e-5f) * g2n[t] + b2n[t];
}

// ---------------- K4: fused edge path (WMMA) ----------------
// Per block: 128 edge rows (fixed b,i; j in [j0, j0+128)), 512 threads = 16 waves.
//   S = edges + q_i * k_j / 16      (bf16, LDS)
//   C1 = relu(S @ Weo^T + beo) + edges ; LN -> Ebuf(f32)/Abuf(bf16)
//   loop hb: H = relu(Abuf @ We1^T[hb] + be1) -> Hbuf ; C3 += H @ We2^T[hb]
//   out = LN(C3 + be2 + Ebuf)
__global__ __launch_bounds__(512) void k_edges(const float* __restrict__ edges,
                                               const float* __restrict__ qb,
                                               const float* __restrict__ kb,
                                               const __bf16* __restrict__ Weo_bf,
                                               const float* __restrict__ beo,
                                               const float* __restrict__ g1e,
                                               const float* __restrict__ b1e,
                                               const __bf16* __restrict__ We1_bf,
                                               const float* __restrict__ be1,
                                               const __bf16* __restrict__ We2_bf,
                                               const float* __restrict__ be2,
                                               const float* __restrict__ g2e,
                                               const float* __restrict__ b2e,
                                               float* __restrict__ out_edges) {
    __shared__ float Ebuf[TM_][257];
    __shared__ __align__(16) __bf16 Abuf[TM_][264];
    __shared__ __align__(16) __bf16 Hbuf[TM_][40];
    __shared__ float qrow[D_];
    __shared__ float psum[TM_][4], psq[TM_][4];
    __shared__ float rowm[TM_], rowr[TM_];

    const int t = threadIdx.x;
    const int lane = t & 31, wid = t >> 5;        // 16 waves
    const size_t row0 = (size_t)blockIdx.x * TM_;
    const int b  = (int)(row0 >> 16);
    const int i  = (int)((row0 >> 8) & 255);
    const int j0 = (int)(row0 & 255);

    if (t < 256) qrow[t] = qb[((size_t)(b * N_ + i)) * D_ + t];
    __syncthreads();

    // stage: Ebuf = edges (residual), Abuf = bf16(edges + q*k/16)
    for (int idx = t; idx < TM_ * 256; idx += 512) {
        const int r = idx >> 8, c = idx & 255;
        const float e = edges[(row0 + r) * E_ + c];
        const float kv = kb[((size_t)(b * N_ + j0 + r)) * D_ + c];
        Ebuf[r][c] = e;
        Abuf[r][c] = (__bf16)(e + qrow[c] * kv * 0.0625f);
    }
    __syncthreads();

    const int rt = wid & 7;       // row tile (GEMM1 & GEMM3): 8 x 16 rows
    const int chalf = wid >> 3;   // column half (GEMM1 & GEMM3)

    // ---- GEMM1: [128x256] x Weo^T[256x256] ----
    {
        v8f acc1[8];
#pragma unroll
        for (int u = 0; u < 8; ++u) acc1[u] = zero_v8f();
        for (int kk = 0; kk < 8; ++kk) {
            v16bf a = load_frag(&Abuf[rt * 16][kk * 32], 264);
            v16bf bmv[8];
#pragma unroll
            for (int cc = 0; cc < 8; ++cc)
                bmv[cc] = load_frag(Weo_bf + (size_t)((chalf * 8 + cc) * 16) * 256 + kk * 32, 256);
#pragma unroll
            for (int cc = 0; cc < 8; ++cc) acc1[cc] = wmma_bf(a, bmv[cc], acc1[cc]);
        }
#pragma unroll
        for (int cc = 0; cc < 8; ++cc) {
            const int col = (chalf * 8 + cc) * 16 + (lane & 15);
            const float bias = beo[col];
#pragma unroll
            for (int v2 = 0; v2 < 8; ++v2) {
                const int r = rt * 16 + ((lane & 16) ? v2 + 8 : v2);
                float x = acc1[cc][v2] + bias;
                x = x > 0.f ? x : 0.f;
                Ebuf[r][col] += x;     // + edges residual
            }
        }
    }
    __syncthreads();

    // ---- LN1 ----
    {
        const int r = t >> 2, s = t & 3;
        float sm = 0.f, sq = 0.f;
        const float* base = &Ebuf[r][s * 64];
        for (int c = 0; c < 64; ++c) {
            const float x = base[c];
            sm += x;
            sq += x * x;
        }
        psum[r][s] = sm;
        psq[r][s] = sq;
        __syncthreads();
        if (s == 0) {
            const float m = (psum[r][0] + psum[r][1] + psum[r][2] + psum[r][3]) * (1.f / 256.f);
            const float q2 = (psq[r][0] + psq[r][1] + psq[r][2] + psq[r][3]) * (1.f / 256.f);
            rowm[r] = m;
            rowr[r] = rsqrtf(q2 - m * m + 1e-5f);
        }
        __syncthreads();
    }
    for (int idx = t; idx < TM_ * 256; idx += 512) {
        const int r = idx >> 8, c = idx & 255;
        const float x = (Ebuf[r][c] - rowm[r]) * rowr[r] * g1e[c] + b1e[c];
        Ebuf[r][c] = x;                 // residual for LN2
        Abuf[r][c] = (__bf16)x;         // A for the MLP
    }
    __syncthreads();

    // ---- edge MLP: hidden in blocks of 32, fused GEMM2 + GEMM3 ----
    v8f acc3[8];
#pragma unroll
    for (int u = 0; u < 8; ++u) acc3[u] = zero_v8f();

    const int rt2 = wid >> 1, ct2 = wid & 1;      // 8 x 2 tiles, one per wave
    for (int hb = 0; hb < 32; ++hb) {
        // GEMM2: preload all 8 B fragments, then 8 back-to-back WMMAs over K
        v16bf b1v[8];
#pragma unroll
        for (int kk = 0; kk < 8; ++kk)
            b1v[kk] = load_frag(We1_bf + (size_t)(hb * 32 + ct2 * 16) * 256 + kk * 32, 256);
        v8f acc2 = zero_v8f();
#pragma unroll
        for (int kk = 0; kk < 8; ++kk) {
            v16bf a = load_frag(&Abuf[rt2 * 16][kk * 32], 264);
            acc2 = wmma_bf(a, b1v[kk], acc2);
        }
        const int hcol = ct2 * 16 + (lane & 15);
        const float bias1 = be1[hb * 32 + hcol];
#pragma unroll
        for (int v2 = 0; v2 < 8; ++v2) {
            const int r = rt2 * 16 + ((lane & 16) ? v2 + 8 : v2);
            const float x = acc2[v2] + bias1;
            Hbuf[r][hcol] = (__bf16)(x > 0.f ? x : 0.f);
        }
        // GEMM3 B fragments: issue loads before the barrier to overlap
        v16bf b3v[8];
#pragma unroll
        for (int cc = 0; cc < 8; ++cc)
            b3v[cc] = load_frag(We2_bf + (size_t)((chalf * 8 + cc) * 16) * 1024 + hb * 32, 1024);
        __syncthreads();
        v16bf a3 = load_frag(&Hbuf[rt * 16][0], 40);
#pragma unroll
        for (int cc = 0; cc < 8; ++cc) acc3[cc] = wmma_bf(a3, b3v[cc], acc3[cc]);
        __syncthreads();
    }

    // ---- epilogue: + be2 + residual, then LN2 -> out ----
#pragma unroll
    for (int cc = 0; cc < 8; ++cc) {
        const int col = (chalf * 8 + cc) * 16 + (lane & 15);
        const float bias = be2[col];
#pragma unroll
        for (int v2 = 0; v2 < 8; ++v2) {
            const int r = rt * 16 + ((lane & 16) ? v2 + 8 : v2);
            Ebuf[r][col] += acc3[cc][v2] + bias;
        }
    }
    __syncthreads();
    {
        const int r = t >> 2, s = t & 3;
        float sm = 0.f, sq = 0.f;
        const float* base = &Ebuf[r][s * 64];
        for (int c = 0; c < 64; ++c) {
            const float x = base[c];
            sm += x;
            sq += x * x;
        }
        psum[r][s] = sm;
        psq[r][s] = sq;
        __syncthreads();
        if (s == 0) {
            const float m = (psum[r][0] + psum[r][1] + psum[r][2] + psum[r][3]) * (1.f / 256.f);
            const float q2 = (psq[r][0] + psq[r][1] + psq[r][2] + psq[r][3]) * (1.f / 256.f);
            rowm[r] = m;
            rowr[r] = rsqrtf(q2 - m * m + 1e-5f);
        }
        __syncthreads();
    }
    float* oe = out_edges + row0 * E_;
    for (int idx = t; idx < TM_ * 256; idx += 512) {
        const int r = idx >> 8, c = idx & 255;
        oe[(size_t)r * E_ + c] = (Ebuf[r][c] - rowm[r]) * rowr[r] * g2e[c] + b2e[c];
    }
}

// ---------------- K5: conds passthrough ----------------
__global__ void k_conds(const float* __restrict__ conds, float* __restrict__ out_conds) {
    const int idx = blockIdx.x * 256 + threadIdx.x;
    if (idx < B_ * D_) out_conds[idx] = conds[idx];
}

extern "C" void kernel_launch(void* const* d_in, const int* in_sizes, int n_in,
                              void* d_out, int out_size, void* d_ws, size_t ws_size,
                              hipStream_t stream) {
    const float* nodes = (const float*)d_in[0];
    const float* edges = (const float*)d_in[1];
    const float* conds = (const float*)d_in[2];
    const float* Wqkv = (const float*)d_in[3];
    const float* bqkv = (const float*)d_in[4];
    const float* Wno = (const float*)d_in[5];
    const float* bno = (const float*)d_in[6];
    const float* Weo = (const float*)d_in[7];
    const float* beo = (const float*)d_in[8];
    const float* g1n = (const float*)d_in[9];
    const float* b1n = (const float*)d_in[10];
    const float* g1e = (const float*)d_in[11];
    const float* b1e = (const float*)d_in[12];
    const float* Wn1 = (const float*)d_in[13];
    const float* bn1 = (const float*)d_in[14];
    const float* Wn2 = (const float*)d_in[15];
    const float* bn2 = (const float*)d_in[16];
    const float* We1 = (const float*)d_in[17];
    const float* be1 = (const float*)d_in[18];
    const float* We2 = (const float*)d_in[19];
    const float* be2 = (const float*)d_in[20];
    const float* g2n = (const float*)d_in[21];
    const float* b2n = (const float*)d_in[22];
    const float* g2e = (const float*)d_in[23];
    const float* b2e = (const float*)d_in[24];

    float* q = (float*)d_ws;
    float* k = q + 131072;
    float* v = k + 131072;
    float* wv = v + 131072;
    __bf16* Weo_bf = (__bf16*)(wv + 131072);
    __bf16* We1_bf = Weo_bf + 65536;
    __bf16* We2_bf = We1_bf + 262144;

    float* out_nodes = (float*)d_out;
    float* out_edges = out_nodes + (size_t)B_ * N_ * D_;
    float* out_conds = out_edges + (size_t)B_ * N_ * N_ * E_;

    k_convert<<<(65536 + 524288) / 256, 256, 0, stream>>>(Weo, We1, We2, Weo_bf, We1_bf, We2_bf);
    k_qkv<<<B_ * N_, 256, 0, stream>>>(nodes, conds, Wqkv, bqkv, q, k, v);
    k_attn<<<B_ * N_, 256, 0, stream>>>(edges, q, k, v, wv);
    k_nodes<<<B_ * N_, 256, 0, stream>>>(wv, nodes, Wno, bno, g1n, b1n, Wn1, bn1, Wn2, bn2, g2n,
                                         b2n, out_nodes);
    k_edges<<<(B_ * N_ * N_) / TM_, 512, 0, stream>>>(edges, q, k, Weo_bf, beo, g1e, b1e, We1_bf,
                                                      be1, We2_bf, be2, g2e, b2e, out_edges);
    k_conds<<<(B_ * D_ + 255) / 256, 256, 0, stream>>>(conds, out_conds);
}
